// FeedForward_80917183856743
// MI455X (gfx1250) — matrix operand order repaired and checked by reading, not compiled, not run on hardware
//
#include <hip/hip_runtime.h>
#include <math.h>

// ---------- types ----------
typedef __attribute__((ext_vector_type(16))) __bf16 v16bf;
typedef __attribute__((ext_vector_type(8)))  float  v8f;
typedef unsigned short u16;
typedef __attribute__((ext_vector_type(4))) u16 u16x4;
typedef __attribute__((ext_vector_type(8))) u16 u16x8;

union FragU {
  u16x8 h[2];
  v16bf v;
};

__device__ __forceinline__ u16 f2bf_bits(float f) {
  unsigned int u = __builtin_bit_cast(unsigned int, f);
  u += 0x7fffu + ((u >> 16) & 1u);          // round-to-nearest-even
  return (u16)(u >> 16);
}

// Branchless erf (Abramowitz-Stegun 7.1.26, |err| < 1.5e-7): v_exp + v_rcp, no
// divergent libm path.
__device__ __forceinline__ float erf_fast(float x) {
  const float ax = fabsf(x);
  const float t  = __builtin_amdgcn_rcpf(fmaf(0.3275911f, ax, 1.0f));
  float y = fmaf(t, 1.061405429f, -1.453152027f);
  y = fmaf(t, y, 1.421413741f);
  y = fmaf(t, y, -0.284496736f);
  y = fmaf(t, y, 0.254829592f);
  const float r = 1.0f - y * t * __expf(-ax * ax);
  return copysignf(r, x);
}

__device__ __forceinline__ float gelu_exact(float v) {
  return 0.5f * v * (1.0f + erf_fast(v * 0.70710678118654752f));
}

// ---------- problem dims (fixed by setup_inputs) ----------
// x: [8192, 1024]  W1: [4096, 1024]  h: [8192, 4096]  W2: [1024, 4096]
static constexpr int Mdim = 8192;
static constexpr int Ddim = 1024;
static constexpr int Hdim = 4096;

// ---------- fp32 -> bf16 conversion of x ----------
__global__ __launch_bounds__(256)
void convert_x_kernel(const float* __restrict__ x, u16* __restrict__ xb) {
  const size_t g = (size_t)blockIdx.x * 256 + threadIdx.x;
  const float4 v = *(const float4*)(x + g * 4);
  u16x4 o;
  o[0] = f2bf_bits(v.x); o[1] = f2bf_bits(v.y);
  o[2] = f2bf_bits(v.z); o[3] = f2bf_bits(v.w);
  *(u16x4*)(xb + g * 4) = o;
}

// ---------- dequant W1: [H=4096][D=1024], C=8, K=256, sub=128 ----------
__global__ __launch_bounds__(256)
void dequant_w1_kernel(const float* __restrict__ cb, const int* __restrict__ idx,
                       u16* __restrict__ W) {
  const int g = blockIdx.x * 256 + threadIdx.x;
  const int e = g * 4;
  const int h = e >> 10;        // / 1024
  const int d = e & 1023;
  const int c = d >> 7;         // / 128
  const int j = d & 127;
  const int code = idx[c * Hdim + h];
  const float4 v = *(const float4*)(cb + ((size_t)(c * 256 + code) << 7) + j);
  u16x4 o;
  o[0] = f2bf_bits(v.x); o[1] = f2bf_bits(v.y);
  o[2] = f2bf_bits(v.z); o[3] = f2bf_bits(v.w);
  *(u16x4*)(W + e) = o;
}

// ---------- dequant W2: [D=1024][H=4096], C=8, K=256, sub=512 ----------
__global__ __launch_bounds__(256)
void dequant_w2_kernel(const float* __restrict__ cb, const int* __restrict__ idx,
                       u16* __restrict__ W) {
  const int g = blockIdx.x * 256 + threadIdx.x;
  const int e = g * 4;
  const int d  = e >> 12;       // / 4096
  const int h2 = e & 4095;
  const int c = h2 >> 9;        // / 512
  const int j = h2 & 511;
  const int code = idx[c * Ddim + d];
  const float4 v = *(const float4*)(cb + ((size_t)(c * 256 + code) << 9) + j);
  u16x4 o;
  o[0] = f2bf_bits(v.x); o[1] = f2bf_bits(v.y);
  o[2] = f2bf_bits(v.z); o[3] = f2bf_bits(v.w);
  *(u16x4*)(W + e) = o;
}

// ---------- async global->LDS tile copy (CDNA5, ASYNCcnt) ----------
// Copies 64 contiguous bytes: offset applies to BOTH the LDS and global
// addresses (ISA 08_async_tensor §4.4), so one LDS-addr VGPR + one global
// offset VGPR + SADDR pair covers all four b128 transfers.
__device__ __forceinline__ void async_copy64(unsigned ldsByte, unsigned gByte,
                                             const u16* saddr) {
  asm volatile(
      "global_load_async_to_lds_b128 %0, %1, %2 offset:0\n\t"
      "global_load_async_to_lds_b128 %0, %1, %2 offset:16\n\t"
      "global_load_async_to_lds_b128 %0, %1, %2 offset:32\n\t"
      "global_load_async_to_lds_b128 %0, %1, %2 offset:48"
      :: "v"(ldsByte), "v"(gByte), "s"(saddr)
      : "memory");
}

__device__ __forceinline__ void wait_async_all() {
  asm volatile("s_wait_asynccnt 0x0" ::: "memory");
}

// ---------- bf16 WMMA GEMM:  C[M,N] = A[M,K] * B[N,K]^T (+bias, epilogue) ----
// Block tile 128x128, K-tile 64 double-buffered in LDS via async copies.
// 8 waves (wave32): each wave owns 32(M) x 64(N) = 2x4 WMMA accumulators.
// EPI == 0 : out = bf16( gelu(acc + bias) )   (feeds GEMM2)
// EPI == 1 : out = f32 ( acc + bias )
template<int EPI>
__global__ __launch_bounds__(256)
void gemm_wmma_bf16(const u16* __restrict__ A, const u16* __restrict__ Bm,
                    const float* __restrict__ bias, void* __restrict__ Cout,
                    int M, int N, int K) {
  constexpr int LDT = 72;                 // 64 + 8 pad -> 144B rows, 16B aligned
  __shared__ u16 As[2][128 * LDT];
  __shared__ u16 Bs[2][128 * LDT];        // 4 * 18432 B = 72 KiB total

  const int tid  = threadIdx.x;
  const int lane = tid & 31;
  const int wave = tid >> 5;
  const int wm   = wave & 3;              // 4 waves along M -> 4*32 = 128
  const int wn   = wave >> 2;             // 2 waves along N -> 2*64 = 128
  const int lr   = lane & 15;
  const int lh   = lane >> 4;

  const int mBlk = blockIdx.y * 128;
  const int nBlk = blockIdx.x * 128;

  const int ldRow = tid >> 1;             // 0..127
  const int ldCol = (tid & 1) * 32;       // 0 or 32 elems (64B halves of a row)

  // per-lane LDS byte addresses for the async-copy destinations (low 32 bits
  // of the generic address == LDS offset, per ISA flat-address truncation)
  const unsigned ldsA0 = (unsigned)(uintptr_t)&As[0][ldRow * LDT + ldCol];
  const unsigned ldsA1 = (unsigned)(uintptr_t)&As[1][ldRow * LDT + ldCol];
  const unsigned ldsB0 = (unsigned)(uintptr_t)&Bs[0][ldRow * LDT + ldCol];
  const unsigned ldsB1 = (unsigned)(uintptr_t)&Bs[1][ldRow * LDT + ldCol];

  // per-lane global byte offsets of this thread's 64B slice at kb==0
  const unsigned gaOff = (unsigned)(((size_t)(mBlk + ldRow) * K + ldCol) * 2);
  const unsigned gbOff = (unsigned)(((size_t)(nBlk + ldRow) * K + ldCol) * 2);

  v8f zero;
  #pragma unroll
  for (int e = 0; e < 8; ++e) zero[e] = 0.0f;
  v8f acc[2][4];
  #pragma unroll
  for (int i = 0; i < 2; ++i)
    #pragma unroll
    for (int j = 0; j < 4; ++j) acc[i][j] = zero;

  // prologue: fill buffer 0
  async_copy64(ldsA0, gaOff, A);
  async_copy64(ldsB0, gbOff, Bm);
  wait_async_all();
  __syncthreads();

  int cur = 0;
  for (int kb = 0; kb < K; kb += 64) {
    // kick off next tile into the other buffer; overlaps with WMMA below
    if (kb + 64 < K) {
      const unsigned kByte = (unsigned)(kb + 64) * 2;
      async_copy64(cur ? ldsA0 : ldsA1, gaOff + kByte, A);
      async_copy64(cur ? ldsB0 : ldsB1, gbOff + kByte, Bm);
    }

    const u16* __restrict__ sA = As[cur];
    const u16* __restrict__ sB = Bs[cur];

    #pragma unroll
    for (int kk = 0; kk < 2; ++kk) {
      const int ko = kk * 32;
      // A fragment (16x32 bf16): lane L -> row M=L%16, K chunks (L/16)*8, +16
      FragU af[2];
      #pragma unroll
      for (int i = 0; i < 2; ++i) {
        const int row = wm * 32 + i * 16 + lr;
        af[i].h[0] = *(const u16x8*)&sA[row * LDT + ko + lh * 8];
        af[i].h[1] = *(const u16x8*)&sA[row * LDT + ko + lh * 8 + 16];
      }
      // B fragment (32x16 bf16): lane L -> col N=L%16, K chunks (L/16)*8, +16
      FragU bfr[4];
      #pragma unroll
      for (int j = 0; j < 4; ++j) {
        const int n = wn * 64 + j * 16 + lr;
        bfr[j].h[0] = *(const u16x8*)&sB[n * LDT + ko + lh * 8];
        bfr[j].h[1] = *(const u16x8*)&sB[n * LDT + ko + lh * 8 + 16];
      }
      #pragma unroll
      for (int i = 0; i < 2; ++i)
        #pragma unroll
        for (int j = 0; j < 4; ++j)
          acc[i][j] = __builtin_amdgcn_wmma_f32_16x16x32_bf16(
              false, af[i].v, false, bfr[j].v, (short)0, acc[i][j],
              false, false);
    }

    // make the freshly copied buffer visible to all waves
    wait_async_all();
    __syncthreads();
    cur ^= 1;
  }

  // Epilogue: C/D layout -> VGPR r holds row (r + 8*lh), col = lr in 16x16 tile
  #pragma unroll
  for (int i = 0; i < 2; ++i) {
    const int m0 = mBlk + wm * 32 + i * 16 + lh * 8;
    #pragma unroll
    for (int j = 0; j < 4; ++j) {
      const int n = nBlk + wn * 64 + j * 16 + lr;
      const float bn = bias[n];
      #pragma unroll
      for (int r = 0; r < 8; ++r) {
        float v = acc[i][j][r] + bn;
        const size_t off = (size_t)(m0 + r) * N + n;
        if (EPI == 0) {
          ((u16*)Cout)[off] = f2bf_bits(gelu_exact(v));
        } else {
          ((float*)Cout)[off] = v;
        }
      }
    }
  }
}

// ---------- launch ----------
extern "C" void kernel_launch(void* const* d_in, const int* in_sizes, int n_in,
                              void* d_out, int out_size, void* d_ws, size_t ws_size,
                              hipStream_t stream) {
  const float* x    = (const float*)d_in[0];
  const float* cb1  = (const float*)d_in[1];
  const int*   idx1 = (const int*)  d_in[2];
  const float* b1   = (const float*)d_in[3];
  const float* cb2  = (const float*)d_in[4];
  const int*   idx2 = (const int*)  d_in[5];
  const float* b2   = (const float*)d_in[6];
  float* out = (float*)d_out;

  // workspace layout (bytes)
  char* ws = (char*)d_ws;
  u16* xb = (u16*)(ws);                              // 8192*1024*2  = 16 MiB
  u16* w1 = (u16*)(ws + (size_t)16777216);           // 4096*1024*2  =  8 MiB
  u16* w2 = (u16*)(ws + (size_t)25165824);           // 1024*4096*2  =  8 MiB
  u16* hb = (u16*)(ws + (size_t)33554432);           // 8192*4096*2  = 64 MiB

  // x -> bf16 : 8,388,608 elems / 4 per thread / 256 per block = 8192 blocks
  convert_x_kernel<<<8192, 256, 0, stream>>>(x, xb);
  // W1 dequant: 4,194,304 elems / 4 / 256 = 4096 blocks
  dequant_w1_kernel<<<4096, 256, 0, stream>>>(cb1, idx1, w1);
  // W2 dequant: 4,194,304 elems / 4 / 256 = 4096 blocks
  dequant_w2_kernel<<<4096, 256, 0, stream>>>(cb2, idx2, w2);

  // GEMM1: [8192,1024] x [4096,1024]^T -> gelu -> bf16 h [8192,4096]
  dim3 g1(Hdim / 128, Mdim / 128);
  gemm_wmma_bf16<0><<<g1, 256, 0, stream>>>(xb, w1, b1, (void*)hb,
                                            Mdim, Hdim, Ddim);
  // GEMM2: [8192,4096] x [1024,4096]^T -> f32 out [8192,1024]
  dim3 g2(Ddim / 128, Mdim / 128);
  gemm_wmma_bf16<1><<<g2, 256, 0, stream>>>(hb, w2, b2, (void*)out,
                                            Mdim, Ddim, Hdim);
}
